// PolicyValueNet_53274774340164
// MI455X (gfx1250) — compile-verified, matrix operand order checked by reading, not run on hardware
//
#include <hip/hip_runtime.h>

// ---------------- CDNA5 WMMA types ----------------
typedef __attribute__((ext_vector_type(16))) __bf16 v16bf;
typedef __attribute__((ext_vector_type(8)))  float  v8f;

union FragBf {
    v16bf v;
    uint4 q[2];
    unsigned short u[16];
};

// ---------------- model constants ----------------
constexpr int BD   = 15;
constexpr int TT   = 225;      // real tokens
constexpr int TP   = 240;      // padded tokens (15 tiles of 16)
constexpr int BB   = 32;       // batch
constexpr int CC   = 256;
constexpr int HH   = 8;
constexpr int HD   = 32;
constexpr int LL   = 12;
constexpr int MLP  = 1024;
constexpr int TBL  = 29 * 29;  // 841
constexpr int ROWS = BB * TP;  // 7680

// ---------------- helpers ----------------
__device__ __forceinline__ unsigned short f2bf(float f) {
    unsigned u = __float_as_uint(f);
    u += 0x7FFFu + ((u >> 16) & 1u);         // round to nearest even
    return (unsigned short)(u >> 16);
}

__device__ __forceinline__ float gelu_t(float x) {
    float x3 = x * x * x;
    return 0.5f * x * (1.f + tanhf(0.7978845608028654f * (x + 0.044715f * x3)));
}

__device__ __forceinline__ float blockReduce256(float v) {
    __shared__ float p[9];
    __syncthreads();                          // protect reuse across calls
    #pragma unroll
    for (int m = 16; m; m >>= 1) v += __shfl_xor(v, m, 32);
    if ((threadIdx.x & 31) == 0) p[threadIdx.x >> 5] = v;
    __syncthreads();
    if (threadIdx.x == 0) {
        float s = 0.f;
        #pragma unroll
        for (int i = 0; i < 8; ++i) s += p[i];
        p[8] = s;
    }
    __syncthreads();
    return p[8];
}

// ---------------- weight convert + transpose: f32 (K,N) -> bf16 (N,K) ----------------
__global__ __launch_bounds__(256) void convT_kernel(const float* __restrict__ W,
                                                    unsigned short* __restrict__ out,
                                                    int K, int N) {
    size_t layer = blockIdx.y;
    W   += layer * (size_t)K * N;
    out += layer * (size_t)K * N;
    int idx = blockIdx.x * 256 + threadIdx.x;
    if (idx < K * N) {
        int kk = idx / N, n = idx % N;
        out[(size_t)n * K + kk] = f2bf(W[idx]);
    }
}

// ---------------- rel-bias expansion (per layer): be[h][t][s], mask folded in ----------------
__global__ __launch_bounds__(256) void bias_expand_kernel(const float* __restrict__ rpb_l,
                                                          float* __restrict__ be) {
    int bid = blockIdx.x;                 // h * TP + t
    int h = bid / TP, t = bid % TP;
    int s = threadIdx.x;
    if (s < TP) {
        float val = -1e30f;
        if (t < TT && s < TT) {
            int dr = t / BD - s / BD + 14;
            int dc = t % BD - s % BD + 14;
            val = rpb_l[(size_t)h * TBL + dr * 29 + dc];
        }
        be[((size_t)h * TP + t) * TP + s] = val;
    }
}

// ---------------- V transpose (per layer): vT[(b,h,hd)][s], s padded to 256 ----------------
__global__ __launch_bounds__(256) void vtrans_kernel(const unsigned short* __restrict__ v,
                                                     unsigned short* __restrict__ vT) {
    int bid = blockIdx.x;                 // ((b*HH + h)*HD + hd)
    int b = bid >> 8;
    int h = (bid >> 5) & 7;
    int hd = bid & 31;
    int s = threadIdx.x;
    if (s < TP)
        vT[(size_t)bid * 256 + s] = v[((size_t)b * TP + s) * CC + h * HD + hd];
}

// ---------------- embedding: x = obs @ te_w + te_b + pos ----------------
__global__ __launch_bounds__(256) void embed_kernel(const float* __restrict__ obs,
                                                    const float* __restrict__ tew,
                                                    const float* __restrict__ teb,
                                                    const float* __restrict__ pos,
                                                    float* __restrict__ x) {
    int bid = blockIdx.x;
    int b = bid / TP, t = bid % TP;
    int c = threadIdx.x;
    float acc = 0.f;
    if (t < TT) {
        acc = teb[c] + pos[t * CC + c];
        #pragma unroll
        for (int f = 0; f < 8; ++f)
            acc += obs[((size_t)b * TT + t) * 8 + f] * tew[f * CC + c];
    }
    x[(size_t)bid * CC + c] = acc;
}

// ---------------- LayerNorm (row of 256), templated output ----------------
template <bool BF16OUT>
__global__ __launch_bounds__(256) void ln_kernel(const float* __restrict__ x,
                                                 const float* __restrict__ sc,
                                                 const float* __restrict__ bi,
                                                 unsigned short* __restrict__ ob,
                                                 float* __restrict__ of) {
    size_t row = blockIdx.x;
    int c = threadIdx.x;
    float v = x[row * CC + c];
    float mean = blockReduce256(v) * (1.f / 256.f);
    float d = v - mean;
    float var = blockReduce256(d * d) * (1.f / 256.f);
    float rs = rsqrtf(var + 1e-6f);
    float y = d * rs * sc[c] + bi[c];
    if (BF16OUT) ob[row * CC + c] = f2bf(y);
    else         of[row * CC + c] = y;
}

// ---------------- generic bf16 WMMA GEMM (128x64 block tile) ----------------
// A: rows x K (bf16, row-major), Bt: N x K (bf16, weights pre-transposed)
// 8 waves, each owning a 32x32 output slab -> 4 WMMAs per 32-K step.
// MODE 1: resout[r][n] += acc + bias       (f32 residual accumulate)
// MODE 2: outb[r][n]    = bf16(acc + bias)
// MODE 3: outb[r][n]    = bf16(gelu(acc + bias))
template <int MODE>
__global__ __launch_bounds__(256) void gemm_bf16(const unsigned short* __restrict__ A,
                                                 const unsigned short* __restrict__ Bt,
                                                 const float* __restrict__ bias,
                                                 float* __restrict__ resout,
                                                 unsigned short* __restrict__ outb,
                                                 int N, int K) {
    __shared__ alignas(16) unsigned short As[128][32];
    __shared__ alignas(16) unsigned short Bs[64][32];
    const int tid  = threadIdx.x;
    const int lane = tid & 31, wave = tid >> 5;
    const int mi = wave & 3, ni = wave >> 2;       // 4 row quads x 2 col slabs
    const size_t row0 = (size_t)blockIdx.x * 128;
    const size_t col0 = (size_t)blockIdx.y * 64;
    v8f c00 = {0.f,0.f,0.f,0.f,0.f,0.f,0.f,0.f};
    v8f c01 = {0.f,0.f,0.f,0.f,0.f,0.f,0.f,0.f};
    v8f c10 = {0.f,0.f,0.f,0.f,0.f,0.f,0.f,0.f};
    v8f c11 = {0.f,0.f,0.f,0.f,0.f,0.f,0.f,0.f};
    const int ar = tid >> 1, ac = (tid & 1) * 16;  // A: 128 rows x 2 half-rows
    const int br = tid >> 2, bc = (tid & 3) * 8;   // B: 64 rows x 4 chunks
    const int mr = lane & 15, koff = (lane >> 4) * 8;

    for (int kb = 0; kb < K; kb += 32) {
        *(uint4*)&As[ar][ac]     = *(const uint4*)&A [(row0 + ar) * K + kb + ac];
        *(uint4*)&As[ar][ac + 8] = *(const uint4*)&A [(row0 + ar) * K + kb + ac + 8];
        *(uint4*)&Bs[br][bc]     = *(const uint4*)&Bt[(col0 + br) * K + kb + bc];
        if (kb + 32 < K) {
            __builtin_prefetch(&A [(row0 + ar) * K + kb + 32 + ac], 0, 0);
            __builtin_prefetch(&Bt[(col0 + br) * K + kb + 32 + bc], 0, 0);
        }
        __syncthreads();
        FragBf fa0, fa1, fb0, fb1;
        fa0.q[0] = *(const uint4*)&As[mi * 32 + mr][koff];
        fa0.q[1] = *(const uint4*)&As[mi * 32 + mr][koff + 16];
        fa1.q[0] = *(const uint4*)&As[mi * 32 + 16 + mr][koff];
        fa1.q[1] = *(const uint4*)&As[mi * 32 + 16 + mr][koff + 16];
        fb0.q[0] = *(const uint4*)&Bs[ni * 32 + mr][koff];
        fb0.q[1] = *(const uint4*)&Bs[ni * 32 + mr][koff + 16];
        fb1.q[0] = *(const uint4*)&Bs[ni * 32 + 16 + mr][koff];
        fb1.q[1] = *(const uint4*)&Bs[ni * 32 + 16 + mr][koff + 16];
        c00 = __builtin_amdgcn_wmma_f32_16x16x32_bf16(false, fa0.v, false, fb0.v,
                                                      (short)0, c00, false, false);
        c01 = __builtin_amdgcn_wmma_f32_16x16x32_bf16(false, fa0.v, false, fb1.v,
                                                      (short)0, c01, false, false);
        c10 = __builtin_amdgcn_wmma_f32_16x16x32_bf16(false, fa1.v, false, fb0.v,
                                                      (short)0, c10, false, false);
        c11 = __builtin_amdgcn_wmma_f32_16x16x32_bf16(false, fa1.v, false, fb1.v,
                                                      (short)0, c11, false, false);
        __syncthreads();
    }
    const int cn = lane & 15, rofs = (lane >> 4) * 8;
    #pragma unroll
    for (int g = 0; g < 8; ++g) {
        size_t r0 = row0 + mi * 32 + rofs + g;
        size_t r1 = r0 + 16;
        size_t cA = col0 + ni * 32 + cn;
        size_t cB = cA + 16;
        float bA = bias[cA], bB = bias[cB];
        float v00 = c00[g] + bA, v01 = c01[g] + bB;
        float v10 = c10[g] + bA, v11 = c11[g] + bB;
        if (MODE == 1) {
            resout[r0 * N + cA] += v00;  resout[r0 * N + cB] += v01;
            resout[r1 * N + cA] += v10;  resout[r1 * N + cB] += v11;
        } else if (MODE == 2) {
            outb[r0 * N + cA] = f2bf(v00);  outb[r0 * N + cB] = f2bf(v01);
            outb[r1 * N + cA] = f2bf(v10);  outb[r1 * N + cB] = f2bf(v11);
        } else {
            outb[r0 * N + cA] = f2bf(gelu_t(v00));  outb[r0 * N + cB] = f2bf(gelu_t(v01));
            outb[r1 * N + cA] = f2bf(gelu_t(v10));  outb[r1 * N + cB] = f2bf(gelu_t(v11));
        }
    }
}

// ---------------- fused attention: one wave per (qtile, head, batch) ----------------
// be: expanded bias [h][t][s] with -1e30 masking folded in
// vT: transposed V [(b,h,hd)][s] with s-stride 256
__global__ __launch_bounds__(32) void attn_kernel(const unsigned short* __restrict__ q,
                                                  const unsigned short* __restrict__ k,
                                                  const unsigned short* __restrict__ vT,
                                                  const float* __restrict__ be,
                                                  unsigned short* __restrict__ ctx) {
    const int ttile = blockIdx.x, h = blockIdx.y, b = blockIdx.z;
    const int lane = threadIdx.x;
    __shared__ float lg[16][240];
    __shared__ alignas(16) unsigned short pr[16][256];
    const float scale = 0.17677669529663687f;  // 1/sqrt(32)
    const int mr = lane & 15;
    const int half = lane >> 4;
    const int koff = half * 8;
    const int rofs = half * 8;

    // Q fragment (16 x 32, K = HD = 32 -> single wmma per key tile)
    FragBf fa;
    {
        size_t qrow = ((size_t)b * TP + ttile * 16 + mr) * CC + h * HD;
        fa.q[0] = *(const uint4*)(q + qrow + koff);
        fa.q[1] = *(const uint4*)(q + qrow + koff + 16);
    }
    // logits = q k^T * scale + bias (masking folded into be)
    const float* berow = be + ((size_t)h * TP + ttile * 16 + rofs) * TP;
    for (int st = 0; st < 15; ++st) {
        FragBf fb;
        size_t krow = ((size_t)b * TP + st * 16 + mr) * CC + h * HD;
        fb.q[0] = *(const uint4*)(k + krow + koff);
        fb.q[1] = *(const uint4*)(k + krow + koff + 16);
        v8f c = {0.f,0.f,0.f,0.f,0.f,0.f,0.f,0.f};
        c = __builtin_amdgcn_wmma_f32_16x16x32_bf16(false, fa.v, false, fb.v,
                                                    (short)0, c, false, false);
        const int col = st * 16 + mr;
        #pragma unroll
        for (int g = 0; g < 8; ++g)
            lg[rofs + g][col] = c[g] * scale + berow[(size_t)g * TP + col];
    }
    __syncthreads();
    // per-row softmax: 2 lanes per row (lane r and r+16 share row r)
    {
        float m = -3e30f;
        for (int s = half; s < TP; s += 2) m = fmaxf(m, lg[mr][s]);
        m = fmaxf(m, __shfl_xor(m, 16, 32));
        float sum = 0.f;
        for (int s = half; s < TP; s += 2) {
            float e = __expf(lg[mr][s] - m);
            sum += e;
            lg[mr][s] = e;
        }
        sum += __shfl_xor(sum, 16, 32);
        float inv = 1.f / sum;
        for (int s = half; s < TP; s += 2) pr[mr][s] = f2bf(lg[mr][s] * inv);
        #pragma unroll
        for (int i = 0; i < 8; ++i) pr[mr][TP + half * 8 + i] = 0;  // pad cols 240..255
    }
    __syncthreads();
    // ctx = probs @ V  (K padded to 256 with zero probs)
    v8f c0 = {0.f,0.f,0.f,0.f,0.f,0.f,0.f,0.f};
    v8f c1 = {0.f,0.f,0.f,0.f,0.f,0.f,0.f,0.f};
    const unsigned short* vrow0 = vT + (((size_t)b * HH + h) * HD + mr) * 256;
    for (int kb = 0; kb < 256; kb += 32) {
        FragBf fp, fv0, fv1;
        fp.q[0]  = *(const uint4*)&pr[mr][kb + koff];
        fp.q[1]  = *(const uint4*)&pr[mr][kb + koff + 16];
        fv0.q[0] = *(const uint4*)(vrow0 + kb + koff);
        fv0.q[1] = *(const uint4*)(vrow0 + kb + koff + 16);
        fv1.q[0] = *(const uint4*)(vrow0 + 16 * 256 + kb + koff);
        fv1.q[1] = *(const uint4*)(vrow0 + 16 * 256 + kb + koff + 16);
        c0 = __builtin_amdgcn_wmma_f32_16x16x32_bf16(false, fp.v, false, fv0.v,
                                                     (short)0, c0, false, false);
        c1 = __builtin_amdgcn_wmma_f32_16x16x32_bf16(false, fp.v, false, fv1.v,
                                                     (short)0, c1, false, false);
    }
    #pragma unroll
    for (int g = 0; g < 8; ++g) {
        int tr = ttile * 16 + g + rofs;
        size_t o = ((size_t)b * TP + tr) * CC + h * HD;
        ctx[o + mr]      = f2bf(c0[g]);
        ctx[o + 16 + mr] = f2bf(c1[g]);
    }
}

// ---------------- policy head: per (b,t) dot(xf_row, pw) + pb ----------------
__global__ __launch_bounds__(256) void policy_kernel(const float* __restrict__ xf,
                                                     const float* __restrict__ pw,
                                                     const float* __restrict__ pb,
                                                     float* __restrict__ out) {
    int bid = blockIdx.x;
    int b = bid / TT, t = bid % TT;
    size_t row = (size_t)b * TP + t;
    float v = xf[row * CC + threadIdx.x] * pw[threadIdx.x];
    float s = blockReduce256(v);
    if (threadIdx.x == 0) out[bid] = s + pb[0];
}

// ---------------- pooling head: attn/mean/max pool -> vf[b][768] ----------------
__global__ __launch_bounds__(256) void pool_kernel(const float* __restrict__ pooled,
                                                   const float* __restrict__ galw,
                                                   const float* __restrict__ galb,
                                                   float* __restrict__ vf) {
    int b = blockIdx.x;
    int tid = threadIdx.x, lane = tid & 31, wave = tid >> 5;
    __shared__ float sl[232];
    __shared__ float sinv;
    for (int t = wave; t < TT; t += 8) {
        float s = 0.f;
        for (int j = lane; j < CC; j += 32)
            s += pooled[((size_t)b * TP + t) * CC + j] * galw[j];
        #pragma unroll
        for (int m = 16; m; m >>= 1) s += __shfl_xor(s, m, 32);
        if (lane == 0) sl[t] = s + galb[0];
    }
    __syncthreads();
    if (wave == 0) {
        float mx = -1e30f;
        for (int t = lane; t < TT; t += 32) mx = fmaxf(mx, sl[t]);
        #pragma unroll
        for (int m = 16; m; m >>= 1) mx = fmaxf(mx, __shfl_xor(mx, m, 32));
        float s = 0.f;
        for (int t = lane; t < TT; t += 32) { float e = __expf(sl[t] - mx); sl[t] = e; s += e; }
        #pragma unroll
        for (int m = 16; m; m >>= 1) s += __shfl_xor(s, m, 32);
        if (lane == 0) sinv = 1.f / s;
    }
    __syncthreads();
    int c = tid;
    float ap = 0.f, mp = 0.f, mx = -1e30f;
    for (int t = 0; t < TT; ++t) {
        float pv = pooled[((size_t)b * TP + t) * CC + c];
        ap += pv * sl[t];
        mp += pv;
        mx = fmaxf(mx, pv);
    }
    vf[(size_t)b * 768 + c]       = ap * sinv;
    vf[(size_t)b * 768 + 256 + c] = mp * (1.f / 225.f);
    vf[(size_t)b * 768 + 512 + c] = mx;
}

// ---------------- small FC (N==256) with gelu ----------------
__global__ __launch_bounds__(256) void fc_kernel(const float* __restrict__ in,
                                                 const float* __restrict__ W,
                                                 const float* __restrict__ bias,
                                                 float* __restrict__ out,
                                                 int K) {
    int b = blockIdx.x, n = threadIdx.x;
    float acc = bias[n];
    for (int kk = 0; kk < K; ++kk)
        acc += in[(size_t)b * K + kk] * W[(size_t)kk * 256 + n];
    out[(size_t)b * 256 + n] = gelu_t(acc);
}

// ---------------- value output: tanh(dot(t2, v1_w) + b) ----------------
__global__ __launch_bounds__(32) void value_kernel(const float* __restrict__ t2,
                                                   const float* __restrict__ w,
                                                   const float* __restrict__ bias,
                                                   float* __restrict__ out) {
    int b = blockIdx.x, lane = threadIdx.x;
    float s = 0.f;
    for (int j = lane; j < 256; j += 32) s += t2[(size_t)b * 256 + j] * w[j];
    #pragma unroll
    for (int m = 16; m; m >>= 1) s += __shfl_xor(s, m, 32);
    if (lane == 0) out[b] = tanhf(s + bias[0]);
}

// ---------------- host orchestration ----------------
extern "C" void kernel_launch(void* const* d_in, const int* in_sizes, int n_in,
                              void* d_out, int out_size, void* d_ws, size_t ws_size,
                              hipStream_t stream) {
    (void)in_sizes; (void)n_in; (void)out_size; (void)ws_size;
    const float* obs   = (const float*)d_in[0];
    const float* te_w  = (const float*)d_in[1];
    const float* te_b  = (const float*)d_in[2];
    const float* pos   = (const float*)d_in[3];
    const float* ln1_s = (const float*)d_in[4];
    const float* ln1_b = (const float*)d_in[5];
    const float* qw    = (const float*)d_in[6];
    const float* qb    = (const float*)d_in[7];
    const float* kw    = (const float*)d_in[8];
    const float* kb    = (const float*)d_in[9];
    const float* vw    = (const float*)d_in[10];
    const float* vb    = (const float*)d_in[11];
    const float* ow    = (const float*)d_in[12];
    const float* ob    = (const float*)d_in[13];
    const float* rpb   = (const float*)d_in[14];
    const float* ln2_s = (const float*)d_in[15];
    const float* ln2_b = (const float*)d_in[16];
    const float* m1w   = (const float*)d_in[17];
    const float* m1b   = (const float*)d_in[18];
    const float* m2w   = (const float*)d_in[19];
    const float* m2b   = (const float*)d_in[20];
    const float* fn_s  = (const float*)d_in[21];
    const float* fn_b  = (const float*)d_in[22];
    const float* pw    = (const float*)d_in[23];
    const float* pb    = (const float*)d_in[24];
    const float* gpn_s = (const float*)d_in[25];
    const float* gpn_b = (const float*)d_in[26];
    const float* gal_w = (const float*)d_in[27];
    const float* gal_b = (const float*)d_in[28];
    const float* gf_w  = (const float*)d_in[29];
    const float* gf_b  = (const float*)d_in[30];
    const float* v0_w  = (const float*)d_in[31];
    const float* v0_b  = (const float*)d_in[32];
    const float* v1_w  = (const float*)d_in[33];
    const float* v1_b  = (const float*)d_in[34];
    float* outp = (float*)d_out;

    // workspace bump allocator (256B aligned)
    char* p = (char*)d_ws;
    auto alloc = [&](size_t bytes) -> char* {
        char* r = p;
        p += (bytes + 255) & ~(size_t)255;
        return r;
    };
    float* x             = (float*)alloc((size_t)ROWS * CC * 4);
    unsigned short* y    = (unsigned short*)alloc((size_t)ROWS * CC * 2);
    unsigned short* qb_  = (unsigned short*)alloc((size_t)(ROWS + 16) * CC * 2);
    unsigned short* kb_  = (unsigned short*)alloc((size_t)(ROWS + 16) * CC * 2);
    unsigned short* vb_  = (unsigned short*)alloc((size_t)(ROWS + 16) * CC * 2);
    unsigned short* vT   = (unsigned short*)alloc((size_t)BB * HH * HD * 256 * 2);
    unsigned short* ctx  = (unsigned short*)alloc((size_t)ROWS * CC * 2);
    unsigned short* h1   = (unsigned short*)alloc((size_t)ROWS * MLP * 2);
    float* be            = (float*)alloc((size_t)HH * TP * TP * 4);
    float* xf            = (float*)alloc((size_t)ROWS * CC * 4);
    float* pooled        = (float*)alloc((size_t)ROWS * CC * 4);
    float* vf            = (float*)alloc((size_t)BB * 768 * 4);
    float* t1            = (float*)alloc((size_t)BB * 256 * 4);
    float* t2            = (float*)alloc((size_t)BB * 256 * 4);
    unsigned short* qwT  = (unsigned short*)alloc((size_t)LL * CC * CC * 2);
    unsigned short* kwT  = (unsigned short*)alloc((size_t)LL * CC * CC * 2);
    unsigned short* vwT  = (unsigned short*)alloc((size_t)LL * CC * CC * 2);
    unsigned short* owT  = (unsigned short*)alloc((size_t)LL * CC * CC * 2);
    unsigned short* m1wT = (unsigned short*)alloc((size_t)LL * CC * MLP * 2);
    unsigned short* m2wT = (unsigned short*)alloc((size_t)LL * MLP * CC * 2);

    // weight convert + transpose (bf16, N-major)
    convT_kernel<<<dim3((CC * CC + 255) / 256, LL), 256, 0, stream>>>(qw, qwT, CC, CC);
    convT_kernel<<<dim3((CC * CC + 255) / 256, LL), 256, 0, stream>>>(kw, kwT, CC, CC);
    convT_kernel<<<dim3((CC * CC + 255) / 256, LL), 256, 0, stream>>>(vw, vwT, CC, CC);
    convT_kernel<<<dim3((CC * CC + 255) / 256, LL), 256, 0, stream>>>(ow, owT, CC, CC);
    convT_kernel<<<dim3((CC * MLP + 255) / 256, LL), 256, 0, stream>>>(m1w, m1wT, CC, MLP);
    convT_kernel<<<dim3((MLP * CC + 255) / 256, LL), 256, 0, stream>>>(m2w, m2wT, MLP, CC);

    // embed
    embed_kernel<<<ROWS, 256, 0, stream>>>(obs, te_w, te_b, pos, x);

    const dim3 gN256(ROWS / 128, CC / 64);
    const dim3 gN1024(ROWS / 128, MLP / 64);

    for (int l = 0; l < LL; ++l) {
        const size_t wC = (size_t)l * CC * CC;
        // LN1 -> y (bf16)
        ln_kernel<true><<<ROWS, 256, 0, stream>>>(x, ln1_s + l * CC, ln1_b + l * CC, y, nullptr);
        // Q, K, V projections (bf16 out)
        gemm_bf16<2><<<gN256, 256, 0, stream>>>(y, qwT + wC, qb + l * CC, nullptr, qb_, CC, CC);
        gemm_bf16<2><<<gN256, 256, 0, stream>>>(y, kwT + wC, kb + l * CC, nullptr, kb_, CC, CC);
        gemm_bf16<2><<<gN256, 256, 0, stream>>>(y, vwT + wC, vb + l * CC, nullptr, vb_, CC, CC);
        // per-layer attention prep: expanded bias + transposed V
        bias_expand_kernel<<<HH * TP, 256, 0, stream>>>(rpb + (size_t)l * HH * TBL, be);
        vtrans_kernel<<<BB * HH * HD, 256, 0, stream>>>(vb_, vT);
        // fused attention
        attn_kernel<<<dim3(15, HH, BB), 32, 0, stream>>>(qb_, kb_, vT, be, ctx);
        // output projection + residual into x
        gemm_bf16<1><<<gN256, 256, 0, stream>>>(ctx, owT + wC, ob + l * CC, x, nullptr, CC, CC);
        // LN2 -> y
        ln_kernel<true><<<ROWS, 256, 0, stream>>>(x, ln2_s + l * CC, ln2_b + l * CC, y, nullptr);
        // MLP: gelu(y @ m1w + b1) -> h1 (bf16); h1 @ m2w + b2 + residual -> x
        gemm_bf16<3><<<gN1024, 256, 0, stream>>>(y, m1wT + (size_t)l * CC * MLP,
                                                 m1b + l * MLP, nullptr, h1, MLP, CC);
        gemm_bf16<1><<<gN256, 256, 0, stream>>>(h1, m2wT + (size_t)l * MLP * CC,
                                                m2b + l * CC, x, nullptr, CC, MLP);
    }

    // heads
    ln_kernel<false><<<ROWS, 256, 0, stream>>>(x, fn_s, fn_b, nullptr, xf);
    ln_kernel<false><<<ROWS, 256, 0, stream>>>(x, gpn_s, gpn_b, nullptr, pooled);

    policy_kernel<<<BB * TT, 256, 0, stream>>>(xf, pw, pb, outp);
    pool_kernel<<<BB, 256, 0, stream>>>(pooled, gal_w, gal_b, vf);
    fc_kernel<<<BB, 256, 0, stream>>>(vf, gf_w, gf_b, t1, 768);
    fc_kernel<<<BB, 256, 0, stream>>>(t1, v0_w, v0_b, t2, 256);
    value_kernel<<<BB, 32, 0, stream>>>(t2, v1_w, v1_b, outp + BB * TT);
}
